// MPNNs2s_738734375372
// MI455X (gfx1250) — matrix operand order repaired and checked by reading, not compiled, not run on hardware
//
#include <hip/hip_runtime.h>
#include <hip/hip_bf16.h>

// ---------------------------------------------------------------------------
// MPNN (NNConv+GRU x3) + Set2Set for gfx1250.
// Core trick: msg = Z @ W2r with Z[e,h*16+k] = out[src(e),h]*eh[e,k],
// W2r[(h,k),o] = nn_w2[(h*64+o),k]  (+64 bias rows folding nn_b2).
// -> one f16 WMMA GEMM per step instead of materializing 537MB of W_e.
// Weight chunks are double-buffered into LDS with async memory->LDS loads
// (ASYNCcnt) when the toolchain exposes them; sync copy fallback otherwise.
// ---------------------------------------------------------------------------

typedef _Float16 half_t;
typedef __attribute__((ext_vector_type(16))) _Float16 v16h;
typedef __attribute__((ext_vector_type(8)))  float    v8f;
typedef __attribute__((ext_vector_type(4)))  int      v4i;

#define NNODES 8192
#define NEDGES 32768
#define NGRAPH 64
#define HDIM   64
#define INC    16
#define NNN    8
#define EHD    16
#define OUTC   12
#define KTOT   1088          // 64*16 + 64 bias rows
#define NCHUNK 34            // KTOT / 32
#define CHUNKH (HDIM * 32)   // halfs per chunk (4KB)

#if defined(__AMDGCN__) && __has_builtin(__builtin_amdgcn_global_load_async_to_lds_b128) && \
    __has_builtin(__builtin_amdgcn_s_wait_asynccnt)
#define ASYNC_LDS 1
typedef __attribute__((address_space(1))) v4i* gas_v4i;
typedef __attribute__((address_space(3))) v4i* las_v4i;
#else
#define ASYNC_LDS 0
#endif

__device__ __forceinline__ float sigf(float x) { return 1.0f / (1.0f + __expf(-x)); }

__device__ __forceinline__ void atomicMaxFloat(float* addr, float val) {
    unsigned int* ua = (unsigned int*)addr;
    unsigned int old = *ua;
    while (__uint_as_float(old) < val) {
        unsigned int assumed = old;
        old = atomicCAS(ua, assumed, __float_as_uint(val));
        if (old == assumed) break;
    }
}

// ---------------------------------------------------------------------------
__global__ void k_zero(float* __restrict__ p, int n) {
    int i = blockIdx.x * blockDim.x + threadIdx.x;
    if (i < n) p[i] = 0.0f;
}

// out0 = relu(x @ lin0_w^T + b)   [N,16]->[N,64]
__global__ void k_lin0(const float* __restrict__ x, const float* __restrict__ w,
                       const float* __restrict__ b, float* __restrict__ out) {
    int idx = blockIdx.x * blockDim.x + threadIdx.x;
    if (idx >= NNODES * HDIM) return;
    int n = idx >> 6, o = idx & 63;
    float acc = b[o];
    #pragma unroll
    for (int i = 0; i < INC; ++i) acc += x[n * INC + i] * w[o * INC + i];
    out[idx] = fmaxf(acc, 0.0f);
}

// eh = relu(edge_attr @ nn_w1^T + b1) stored f16   [E,8]->[E,16]
__global__ void k_eh(const float* __restrict__ ea, const float* __restrict__ w,
                     const float* __restrict__ b, half_t* __restrict__ eh) {
    int idx = blockIdx.x * blockDim.x + threadIdx.x;
    if (idx >= NEDGES * EHD) return;
    int e = idx >> 4, k = idx & 15;
    float acc = b[k];
    #pragma unroll
    for (int i = 0; i < NNN; ++i) acc += ea[e * NNN + i] * w[k * NNN + i];
    eh[idx] = (half_t)fmaxf(acc, 0.0f);
}

// Repack nn_w2 [4096,16] (+nn_b2[4096]) -> f16 [chunk=34][o=64][klocal=32]
__global__ void k_w2pack(const float* __restrict__ w2, const float* __restrict__ b2,
                         half_t* __restrict__ dst) {
    int idx = blockIdx.x * blockDim.x + threadIdx.x;   // K*64 + o
    if (idx >= KTOT * HDIM) return;
    int K = idx >> 6, o = idx & 63;
    float v;
    if (K < 1024) { int h = K >> 4, k = K & 15; v = w2[(h * HDIM + o) * EHD + k]; }
    else          { v = b2[(K - 1024) * HDIM + o]; }
    int chunk = K >> 5, kl = K & 31;
    dst[(chunk * HDIM + o) * 32 + kl] = (half_t)v;
}

#if ASYNC_LDS
// Stage one 4KB weight chunk memory->LDS with async b128 loads (128 threads).
__device__ __forceinline__ void stage_chunk_async(const half_t* __restrict__ gsrc,
                                                  half_t* lds) {
    const char* g = (const char*)gsrc;
    char* l = (char*)lds;
    int t = threadIdx.x;
    __builtin_amdgcn_global_load_async_to_lds_b128(
        (gas_v4i)(g + t * 16), (las_v4i)(l + t * 16), 0, 0);
    __builtin_amdgcn_global_load_async_to_lds_b128(
        (gas_v4i)(g + 2048 + t * 16), (las_v4i)(l + 2048 + t * 16), 0, 0);
}
#endif

// Fused message GEMM + segment_sum:  agg[dst(e),:] += out[src(e),:] x W_e.
// Block = 128 threads = 4 waves; wave w handles edge tile blockIdx*4+w (16 edges).
__global__ __launch_bounds__(128) void k_msg(const float* __restrict__ feat,
                                             const half_t* __restrict__ eh,
                                             const half_t* __restrict__ w2p,
                                             const int* __restrict__ src,
                                             const int* __restrict__ dst,
                                             float* __restrict__ agg) {
#if ASYNC_LDS
    __shared__ __attribute__((aligned(32))) half_t sB[2][CHUNKH];   // double buffer
#else
    __shared__ __attribute__((aligned(32))) half_t sB[1][CHUNKH];
#endif
    __shared__ half_t sA[4][16 * HDIM];                             // [edge][h]
    __shared__ half_t sE[4][16 * EHD];                              // [edge][k]
    __shared__ int    sD[4][16];

    const int lane = threadIdx.x & 31;
    const int wave = threadIdx.x >> 5;
    const int e0   = (blockIdx.x * 4 + wave) * 16;

    if (lane < 16) sD[wave][lane] = dst[e0 + lane];
    for (int idx = lane; idx < 16 * HDIM; idx += 32) {
        int e = idx >> 6, h = idx & 63;
        sA[wave][idx] = (half_t)feat[src[e0 + e] * HDIM + h];
    }
    for (int idx = lane; idx < 16 * EHD; idx += 32)
        sE[wave][idx] = eh[e0 * EHD + idx];

    const int abase = (lane < 16) ? 0 : 16;     // A-fragment K sub-block
    const int ae    = lane & 15;                // edge row for A build
    const int bn    = lane & 15;                // N (o) column for B
    const int bko   = (lane < 16) ? 0 : 16;     // B K offset

#if ASYNC_LDS
    stage_chunk_async(w2p, sB[0]);
    __builtin_amdgcn_s_wait_asynccnt(0);
    __syncthreads();
#endif

    v8f acc[4] = {};

    for (int ch = 0; ch < NCHUNK; ++ch) {
#if ASYNC_LDS
        if (ch + 1 < NCHUNK)
            stage_chunk_async(w2p + (ch + 1) * CHUNKH, sB[(ch + 1) & 1]);
        const half_t* Bc = sB[ch & 1];
#else
        __syncthreads();
        {
            const uint4* g4 = (const uint4*)(w2p + ch * CHUNKH);
            uint4* s4 = (uint4*)sB[0];
            for (int i = threadIdx.x; i < 256; i += 128) s4[i] = g4[i];
        }
        if (ch + 1 < NCHUNK) __builtin_prefetch(w2p + (ch + 1) * CHUNKH, 0, 3);
        __syncthreads();
        const half_t* Bc = sB[0];
#endif
        // Build A fragment: chunks 0..31 -> Z = out_src[h]*eh[k]; 32..33 -> bias rows.
        v16h a;
        if (ch < 32) {
            #pragma unroll
            for (int i = 0; i < 16; ++i) {
                int kl = abase + i;
                int Kg = ch * 32 + kl;
                a[i] = sA[wave][ae * HDIM + (Kg >> 4)] * sE[wave][ae * EHD + (Kg & 15)];
            }
        } else {
            #pragma unroll
            for (int i = 0; i < 16; ++i) {
                int kl = abase + i;
                a[i] = sA[wave][ae * HDIM + (ch - 32) * 32 + kl];
            }
        }
        #pragma unroll
        for (int t = 0; t < 4; ++t) {
            v16h b = *(const v16h*)(Bc + (t * 16 + bn) * 32 + bko);
            acc[t] = __builtin_amdgcn_wmma_f32_16x16x32_f16(
                false, a, false, b, (short)0, acc[t], false, false);
        }
#if ASYNC_LDS
        if (ch + 1 < NCHUNK) {
            __builtin_amdgcn_s_wait_asynccnt(0);
            __syncthreads();
        }
#endif
    }

    // C layout: VGPR r -> M=r (lanes 0-15) / M=r+8 (lanes 16-31), N = lane&15.
    const int eoff = (lane < 16) ? 0 : 8;
    #pragma unroll
    for (int r = 0; r < 8; ++r) {
        int d = sD[wave][r + eoff];
        #pragma unroll
        for (int t = 0; t < 4; ++t)
            atomicAdd(&agg[d * HDIM + t * 16 + (lane & 15)], acc[t][r]);
    }
}

// m = relu(agg + out @ conv_root + conv_bias)
__global__ void k_m(const float* __restrict__ feat, const float* __restrict__ agg,
                    const float* __restrict__ root, const float* __restrict__ bias,
                    float* __restrict__ m) {
    int idx = blockIdx.x * blockDim.x + threadIdx.x;
    if (idx >= NNODES * HDIM) return;
    int n = idx >> 6, o = idx & 63;
    float acc = agg[idx] + bias[o];
    const float* fr = feat + n * HDIM;
    for (int h = 0; h < HDIM; ++h) acc += fr[h] * root[h * HDIM + o];
    m[idx] = fmaxf(acc, 0.0f);
}

// GRU cell (h_out separate buffer; h == out entering each step)
__global__ void k_gru(const float* __restrict__ m, const float* __restrict__ h,
                      const float* __restrict__ wih, const float* __restrict__ whh,
                      const float* __restrict__ bih, const float* __restrict__ bhh,
                      float* __restrict__ hout) {
    int idx = blockIdx.x * blockDim.x + threadIdx.x;
    if (idx >= NNODES * HDIM) return;
    int n = idx >> 6, j = idx & 63;
    const float* mr = m + n * HDIM;
    const float* hr = h + n * HDIM;
    float gr = bih[j], gz = bih[64 + j], gn = bih[128 + j];
    float qr = bhh[j], qz = bhh[64 + j], qn = bhh[128 + j];
    for (int i = 0; i < HDIM; ++i) {
        float mi = mr[i], hi = hr[i];
        gr += mi * wih[j * HDIM + i];
        gz += mi * wih[(64 + j) * HDIM + i];
        gn += mi * wih[(128 + j) * HDIM + i];
        qr += hi * whh[j * HDIM + i];
        qz += hi * whh[(64 + j) * HDIM + i];
        qn += hi * whh[(128 + j) * HDIM + i];
    }
    float r = sigf(gr + qr), z = sigf(gz + qz);
    float ng = tanhf(gn + r * qn);
    hout[idx] = (1.0f - z) * ng + z * hr[j];
}

// LSTM cell: one block per graph (G=64), 256 threads = 4H gates
__global__ __launch_bounds__(256) void k_lstm(const float* __restrict__ inp, int in_d,
                                              const float* __restrict__ wih,
                                              const float* __restrict__ whh,
                                              const float* __restrict__ bih,
                                              const float* __restrict__ bhh,
                                              float* __restrict__ h, float* __restrict__ c) {
    __shared__ float sIn[128];
    __shared__ float sH[64];
    __shared__ float sG[256];
    int g = blockIdx.x, t = threadIdx.x;
    if (t < in_d) sIn[t] = inp[g * in_d + t];
    if (t < 64)   sH[t]  = h[g * 64 + t];
    __syncthreads();
    float acc = bih[t] + bhh[t];
    for (int i = 0; i < in_d; ++i) acc += sIn[i] * wih[t * in_d + i];
    for (int i = 0; i < 64; ++i)   acc += sH[i] * whh[t * 64 + i];
    sG[t] = acc;
    __syncthreads();
    if (t < 64) {
        float ig = sG[t], fg = sG[64 + t], gg = sG[128 + t], og = sG[192 + t];
        float c2 = sigf(fg) * c[g * 64 + t] + sigf(ig) * tanhf(gg);
        float h2 = sigf(og) * tanhf(c2);
        c[g * 64 + t] = c2;
        h[g * 64 + t] = h2;
    }
}

__global__ void k_init_att(float* __restrict__ r, float* __restrict__ emax,
                           float* __restrict__ asum) {
    int i = blockIdx.x * blockDim.x + threadIdx.x;
    if (i < NGRAPH * HDIM) r[i] = 0.0f;
    else if (i < NGRAPH * HDIM + NGRAPH) emax[i - NGRAPH * HDIM] = -3.0e38f;
    else if (i < NGRAPH * HDIM + 2 * NGRAPH) asum[i - NGRAPH * HDIM - NGRAPH] = 0.0f;
}

__global__ void k_e(const float* __restrict__ feat, const float* __restrict__ q,
                    const int* __restrict__ batch, float* __restrict__ e,
                    float* __restrict__ emax) {
    int n = blockIdx.x * blockDim.x + threadIdx.x;
    if (n >= NNODES) return;
    int g = batch[n];
    float acc = 0.0f;
    const float* fr = feat + n * HDIM;
    const float* qr = q + g * HDIM;
    for (int i = 0; i < HDIM; ++i) acc += fr[i] * qr[i];
    e[n] = acc;
    atomicMaxFloat(&emax[g], acc);
}

__global__ void k_exp(const float* __restrict__ e, const int* __restrict__ batch,
                      const float* __restrict__ emax, float* __restrict__ a,
                      float* __restrict__ asum) {
    int n = blockIdx.x * blockDim.x + threadIdx.x;
    if (n >= NNODES) return;
    int g = batch[n];
    float v = __expf(e[n] - emax[g]);
    a[n] = v;
    atomicAdd(&asum[g], v);
}

__global__ void k_r(const float* __restrict__ a, const float* __restrict__ asum,
                    const int* __restrict__ batch, const float* __restrict__ feat,
                    float* __restrict__ r) {
    int idx = blockIdx.x * blockDim.x + threadIdx.x;
    if (idx >= NNODES * HDIM) return;
    int n = idx >> 6, h = idx & 63;
    int g = batch[n];
    float w = a[n] / (asum[g] + 1e-16f);
    atomicAdd(&r[g * HDIM + h], w * feat[idx]);
}

__global__ void k_qstar(const float* __restrict__ q, const float* __restrict__ r,
                        float* __restrict__ qs) {
    int idx = blockIdx.x * blockDim.x + threadIdx.x;
    if (idx >= NGRAPH * 2 * HDIM) return;
    int g = idx >> 7, j = idx & 127;
    qs[idx] = (j < HDIM) ? q[g * HDIM + j] : r[g * HDIM + j - HDIM];
}

__global__ void k_lin1(const float* __restrict__ qs, const float* __restrict__ w,
                       const float* __restrict__ b, float* __restrict__ o1) {
    int idx = blockIdx.x * blockDim.x + threadIdx.x;
    if (idx >= NGRAPH * HDIM) return;
    int g = idx >> 6, j = idx & 63;
    float acc = b[j];
    for (int i = 0; i < 2 * HDIM; ++i) acc += qs[g * 2 * HDIM + i] * w[j * 2 * HDIM + i];
    o1[idx] = fmaxf(acc, 0.0f);
}

__global__ void k_lin2(const float* __restrict__ o1, const float* __restrict__ w,
                       const float* __restrict__ b, float* __restrict__ out) {
    int idx = blockIdx.x * blockDim.x + threadIdx.x;
    if (idx >= NGRAPH * OUTC) return;
    int g = idx / OUTC, j = idx % OUTC;
    float acc = b[j];
    for (int i = 0; i < HDIM; ++i) acc += o1[g * HDIM + i] * w[j * HDIM + i];
    out[idx] = acc;
}

// ---------------------------------------------------------------------------
extern "C" void kernel_launch(void* const* d_in, const int* in_sizes, int n_in,
                              void* d_out, int out_size, void* d_ws, size_t ws_size,
                              hipStream_t stream) {
    const float* x        = (const float*)d_in[0];
    const int*   ei       = (const int*)d_in[1];
    const float* ea       = (const float*)d_in[2];
    const int*   batch    = (const int*)d_in[3];
    const float* lin0_w   = (const float*)d_in[4];
    const float* lin0_b   = (const float*)d_in[5];
    const float* nn_w1    = (const float*)d_in[6];
    const float* nn_b1    = (const float*)d_in[7];
    const float* nn_w2    = (const float*)d_in[8];
    const float* nn_b2    = (const float*)d_in[9];
    const float* convroot = (const float*)d_in[10];
    const float* convbias = (const float*)d_in[11];
    const float* gru_wih  = (const float*)d_in[12];
    const float* gru_whh  = (const float*)d_in[13];
    const float* gru_bih  = (const float*)d_in[14];
    const float* gru_bhh  = (const float*)d_in[15];
    const float* lstm_wih[3] = {(const float*)d_in[16], (const float*)d_in[20], (const float*)d_in[24]};
    const float* lstm_whh[3] = {(const float*)d_in[17], (const float*)d_in[21], (const float*)d_in[25]};
    const float* lstm_bih[3] = {(const float*)d_in[18], (const float*)d_in[22], (const float*)d_in[26]};
    const float* lstm_bhh[3] = {(const float*)d_in[19], (const float*)d_in[23], (const float*)d_in[27]};
    const float* lin1_w   = (const float*)d_in[28];
    const float* lin1_b   = (const float*)d_in[29];
    const float* lin2_w   = (const float*)d_in[30];
    const float* lin2_b   = (const float*)d_in[31];
    float* out = (float*)d_out;

    const int* src = ei;
    const int* dst = ei + NEDGES;

    // workspace carve-up
    float* bufA   = (float*)d_ws;
    float* bufB   = bufA + NNODES * HDIM;
    float* m_buf  = bufB + NNODES * HDIM;
    float* agg    = m_buf + NNODES * HDIM;
    float* e_buf  = agg + NNODES * HDIM;
    float* a_buf  = e_buf + NNODES;
    float* emax   = a_buf + NNODES;
    float* asum   = emax + NGRAPH;
    float* r_buf  = asum + NGRAPH;
    float* hs     = r_buf + NGRAPH * HDIM;           // 3 * G*H
    float* cs     = hs + 3 * NGRAPH * HDIM;          // 3 * G*H
    float* q_star = cs + 3 * NGRAPH * HDIM;          // G*2H
    float* o1     = q_star + NGRAPH * 2 * HDIM;      // G*H
    half_t* eh16  = (half_t*)(o1 + NGRAPH * HDIM);   // E*16
    half_t* w2p   = eh16 + (size_t)NEDGES * EHD;     // 34*64*32

    const int T = 256;
    const int nNH = NNODES * HDIM;

    k_lin0<<<nNH / T, T, 0, stream>>>(x, lin0_w, lin0_b, bufA);
    k_eh<<<(NEDGES * EHD) / T, T, 0, stream>>>(ea, nn_w1, nn_b1, eh16);
    k_w2pack<<<(KTOT * HDIM) / T, T, 0, stream>>>(nn_w2, nn_b2, w2p);

    float* cur = bufA;
    float* nxt = bufB;
    for (int step = 0; step < 3; ++step) {
        k_zero<<<nNH / T, T, 0, stream>>>(agg, nNH);
        k_msg<<<NEDGES / 64, 128, 0, stream>>>(cur, eh16, w2p, src, dst, agg);
        k_m<<<nNH / T, T, 0, stream>>>(cur, agg, convroot, convbias, m_buf);
        k_gru<<<nNH / T, T, 0, stream>>>(m_buf, cur, gru_wih, gru_whh, gru_bih, gru_bhh, nxt);
        float* tmp = cur; cur = nxt; nxt = tmp;
    }

    // Set2Set: zero hs(3)+cs(3)+q_star contiguously
    k_zero<<<(6 * NGRAPH * HDIM + 2 * NGRAPH * HDIM + T - 1) / T, T, 0, stream>>>(
        hs, 6 * NGRAPH * HDIM + NGRAPH * 2 * HDIM);
    for (int s = 0; s < 3; ++s) {
        for (int l = 0; l < 3; ++l) {
            const float* inp = (l == 0) ? q_star : (hs + (l - 1) * NGRAPH * HDIM);
            int in_d = (l == 0) ? 2 * HDIM : HDIM;
            k_lstm<<<NGRAPH, 256, 0, stream>>>(inp, in_d, lstm_wih[l], lstm_whh[l],
                                               lstm_bih[l], lstm_bhh[l],
                                               hs + l * NGRAPH * HDIM,
                                               cs + l * NGRAPH * HDIM);
        }
        const float* q = hs + 2 * NGRAPH * HDIM;
        k_init_att<<<(NGRAPH * HDIM + 2 * NGRAPH + T - 1) / T, T, 0, stream>>>(r_buf, emax, asum);
        k_e<<<NNODES / T, T, 0, stream>>>(cur, q, batch, e_buf, emax);
        k_exp<<<NNODES / T, T, 0, stream>>>(e_buf, batch, emax, a_buf, asum);
        k_r<<<nNH / T, T, 0, stream>>>(a_buf, asum, batch, cur, r_buf);
        k_qstar<<<(NGRAPH * 2 * HDIM) / T, T, 0, stream>>>(q, r_buf, q_star);
    }

    k_lin1<<<(NGRAPH * HDIM) / T, T, 0, stream>>>(q_star, lin1_w, lin1_b, o1);
    k_lin2<<<(NGRAPH * OUTC + T - 1) / T, T, 0, stream>>>(o1, lin2_w, lin2_b, out);
}
